// GraphLaplacianAttention_53128745452151
// MI455X (gfx1250) — matrix operand order repaired and checked by reading, not compiled, not run on hardware
//
#include <hip/hip_runtime.h>
#include <math.h>

// ---------------------------------------------------------------------------
// GraphLaplacianAttention for MI455X (gfx1250, wave32, WMMA f32_16x16x32_f16)
// ---------------------------------------------------------------------------

typedef __attribute__((ext_vector_type(16))) _Float16 v16h;
typedef __attribute__((ext_vector_type(8)))  _Float16 v8h;
typedef __attribute__((ext_vector_type(8)))  float    v8f;

#define NODES 16384   // B*NPG
#define DIMN  512
#define BG    64
#define NPGN  256
#define HN    8
#define DH    64
#define EHN   16
#define EN    2048
#define SCALEF 0.125f // 64^-0.5

__device__ __forceinline__ v8f wmma16(v16h a, v16h b, v8f c) {
  return __builtin_amdgcn_wmma_f32_16x16x32_f16(
      /*neg_a=*/false, a, /*neg_b=*/false, b,
      /*c_mod=*/(short)0, c, /*reuse_a=*/false, /*reuse_b=*/false);
}

// A fragment: 16x32 f16, row-major memory mem[row*ld + k].
// Per ISA: lane L holds row (L&15); element e maps to
// k = kbase + (e>>3)*16 + (L>>4)*8 + (e&7)  -> two contiguous 16B chunks.
__device__ __forceinline__ v16h load_a16(const _Float16* __restrict__ rowbase,
                                         int ld, int kbase, int lane) {
  const int m = lane & 15, half = lane >> 4;
  const _Float16* p = rowbase + m * ld + kbase + half * 8;
  v8h lo = *(const v8h*)(p);
  v8h hi = *(const v8h*)(p + 16);
  v16h f;
#pragma unroll
  for (int i = 0; i < 8; ++i) { f[i] = lo[i]; f[i + 8] = hi[i]; }
  return f;
}

// A fragment, but source is f32 (converted to f16 in-register).
__device__ __forceinline__ v16h load_a32(const float* __restrict__ rowbase,
                                         int ld, int kbase, int lane) {
  const int m = lane & 15, half = lane >> 4;
  const float* p = rowbase + m * ld + kbase + half * 8;
  v16h f;
#pragma unroll
  for (int i = 0; i < 8; ++i) {
    f[i]     = (_Float16)p[i];
    f[i + 8] = (_Float16)p[16 + i];
  }
  return f;
}

// B fragment: 32x16 f16 (K x N), memory is n-major: mem[n*ld + k].
// Per ISA: lane L holds column (L&15); k = kbase + (e>>3)*16 + (L>>4)*8 + (e&7).
__device__ __forceinline__ v16h load_b16(const _Float16* __restrict__ colbase,
                                         int ld, int kbase, int lane) {
  const int n = lane & 15, half = lane >> 4;
  const _Float16* p = colbase + n * ld + kbase + half * 8;
  v8h lo = *(const v8h*)(p);
  v8h hi = *(const v8h*)(p + 16);
  v16h f;
#pragma unroll
  for (int i = 0; i < 8; ++i) { f[i] = lo[i]; f[i + 8] = hi[i]; }
  return f;
}

// ---------------------------------------------------------------------------
// 0) f32 -> f16 conversion
// ---------------------------------------------------------------------------
__global__ void gla_cvt_f16(const float* __restrict__ src,
                            _Float16* __restrict__ dst, int n) {
  int i = blockIdx.x * 256 + threadIdx.x;
  if (i < n) dst[i] = (_Float16)src[i];
}

// ---------------------------------------------------------------------------
// 1) QKV projection: y = x @ W^T, scatter into [b,h,n,d] layouts.
//    grid = (1024 node-tiles, 32 channel-tiles, 3 {q,k,v}), block = 1 wave.
// ---------------------------------------------------------------------------
__global__ void gla_qkv_gemm(const _Float16* __restrict__ xb,
                             const _Float16* __restrict__ Wq16,
                             const _Float16* __restrict__ Wk16,
                             const _Float16* __restrict__ Wv16,
                             _Float16* __restrict__ qb,
                             _Float16* __restrict__ kbuf,
                             _Float16* __restrict__ vT,
                             float* __restrict__ vf) {
  const int lane = threadIdx.x;
  const int tm = blockIdx.x, tc = blockIdx.y, z = blockIdx.z;
  const _Float16* W = (z == 0) ? Wq16 : (z == 1) ? Wk16 : Wv16;
  const _Float16* arow = xb + (size_t)tm * 16 * DIMN;
  const _Float16* bcol = W + (size_t)tc * 16 * DIMN;   // W[c][k], k contiguous
  v8f acc = {};
#pragma unroll
  for (int k0 = 0; k0 < DIMN; k0 += 32)
    acc = wmma16(load_a16(arow, DIMN, k0, lane), load_b16(bcol, DIMN, k0, lane), acc);

  const float scale = (z == 1) ? SCALEF : 1.0f;
  const int half = lane >> 4;
  const int cc = tc * 16 + (lane & 15);
  const int h = cc >> 6, d = cc & 63;
#pragma unroll
  for (int r = 0; r < 8; ++r) {
    const int node = tm * 16 + r + half * 8;
    const int b = node >> 8, np = node & 255;
    const int bh = b * HN + h;
    const float val = acc[r] * scale;
    if (z == 0) {
      qb[((size_t)bh * NPGN + np) * DH + d] = (_Float16)val;
    } else if (z == 1) {
      kbuf[((size_t)bh * NPGN + np) * DH + d] = (_Float16)val;
    } else {
      vT[((size_t)bh * DH + d) * NPGN + np] = (_Float16)val;  // [b,h,d,n] for B-frags
      vf[((size_t)bh * NPGN + np) * DH + d] = val;            // f32 for residual
    }
  }
}

// ---------------------------------------------------------------------------
// 2) Scores for all 8 heads + fused head-expand -> GELU -> squeeze.
//    Writes pre-softmax mixed logits P[b,h,n,m].
//    grid = (16 m-tiles, 16 n-tiles, 64 graphs), block = 1 wave.
// ---------------------------------------------------------------------------
__global__ void gla_scores(const _Float16* __restrict__ qb,
                           const _Float16* __restrict__ kbuf,
                           const float* __restrict__ We,
                           const float* __restrict__ Ws,
                           float* __restrict__ P) {
  __shared__ float sWe[EHN * HN];
  __shared__ float sWs[HN * EHN];
  const int lane = threadIdx.x;
  for (int i = lane; i < EHN * HN; i += 32) { sWe[i] = We[i]; sWs[i] = Ws[i]; }
  __syncthreads();

  const int tmm = blockIdx.x, tnn = blockIdx.y, b = blockIdx.z;
  v8f zero = {};
  v8f acc[HN];
#pragma unroll
  for (int h = 0; h < HN; ++h) acc[h] = zero;

#pragma unroll
  for (int h = 0; h < HN; ++h) {
    const _Float16* qbh = qb   + ((size_t)(b * HN + h) * NPGN + tnn * 16) * DH;
    const _Float16* kbh = kbuf + ((size_t)(b * HN + h) * NPGN + tmm * 16) * DH;
    acc[h] = wmma16(load_a16(qbh, DH, 0, lane),  load_b16(kbh, DH, 0, lane),  acc[h]);
    acc[h] = wmma16(load_a16(qbh, DH, 32, lane), load_b16(kbh, DH, 32, lane), acc[h]);
  }

  const int half = lane >> 4;
  const int m = tmm * 16 + (lane & 15);
#pragma unroll
  for (int r = 0; r < 8; ++r) {
    const int n = tnn * 16 + r + half * 8;
    float s[HN];
#pragma unroll
    for (int h = 0; h < HN; ++h) s[h] = acc[h][r];
    float g[EHN];
#pragma unroll
    for (int e = 0; e < EHN; ++e) {
      float t = 0.f;
#pragma unroll
      for (int h = 0; h < HN; ++h) t += sWe[e * HN + h] * s[h];
      g[e] = 0.5f * t * (1.0f + erff(t * 0.70710678118654752f));  // exact GELU
    }
#pragma unroll
    for (int h = 0; h < HN; ++h) {
      float t = 0.f;
#pragma unroll
      for (int e = 0; e < EHN; ++e) t += sWs[h * EHN + e] * g[e];
      P[((size_t)(b * HN + h) * NPGN + n) * NPGN + m] = t;
    }
  }
}

// ---------------------------------------------------------------------------
// 3) Row softmax over m (256 elems/row), in place. One wave per row.
//    block = (32,8), grid = 131072/8.
// ---------------------------------------------------------------------------
__global__ void gla_softmax(float* __restrict__ P) {
  const int lane = threadIdx.x;
  const size_t row = (size_t)blockIdx.x * 8 + threadIdx.y;
  float* p = P + row * NPGN + lane * 8;
  float v[8];
#pragma unroll
  for (int i = 0; i < 8; ++i) v[i] = p[i];
  float mx = v[0];
#pragma unroll
  for (int i = 1; i < 8; ++i) mx = fmaxf(mx, v[i]);
#pragma unroll
  for (int off = 16; off >= 1; off >>= 1) mx = fmaxf(mx, __shfl_xor(mx, off, 32));
  float sum = 0.f;
#pragma unroll
  for (int i = 0; i < 8; ++i) { v[i] = __expf(v[i] - mx); sum += v[i]; }
#pragma unroll
  for (int off = 16; off >= 1; off >>= 1) sum += __shfl_xor(sum, off, 32);
  const float inv = 1.0f / sum;
#pragma unroll
  for (int i = 0; i < 8; ++i) p[i] = v[i] * inv;
}

// ---------------------------------------------------------------------------
// 4) out = v - A @ v per (b,h); writes f16 [node][c] for the projection GEMM.
//    grid = (4 d-tiles, 16 n-tiles, 512 bh), block = 1 wave.
// ---------------------------------------------------------------------------
__global__ void gla_av(const float* __restrict__ P,
                       const _Float16* __restrict__ vT,
                       const float* __restrict__ vf,
                       _Float16* __restrict__ outs) {
  const int lane = threadIdx.x;
  const int td = blockIdx.x, tn = blockIdx.y, bh = blockIdx.z;
  const float*    arow = P  + (size_t)bh * NPGN * NPGN + (size_t)tn * 16 * NPGN;
  const _Float16* bcol = vT + (size_t)bh * NPGN * DH   + (size_t)td * 16 * NPGN;
  v8f acc = {};
#pragma unroll
  for (int k0 = 0; k0 < NPGN; k0 += 32)
    acc = wmma16(load_a32(arow, NPGN, k0, lane), load_b16(bcol, NPGN, k0, lane), acc);

  const int half = lane >> 4;
  const int d = td * 16 + (lane & 15);
  const int b = bh >> 3, h = bh & 7;
#pragma unroll
  for (int r = 0; r < 8; ++r) {
    const int n = tn * 16 + r + half * 8;
    const float val = vf[((size_t)bh * NPGN + n) * DH + d] - acc[r];
    const int node = b * NPGN + n;
    outs[(size_t)node * DIMN + h * DH + d] = (_Float16)val;
  }
}

// ---------------------------------------------------------------------------
// 5) Output projection: y = outs @ Wo^T + bo  (f32 to d_out).
//    grid = (1024, 32), block = 1 wave.
// ---------------------------------------------------------------------------
__global__ void gla_proj(const _Float16* __restrict__ outs,
                         const _Float16* __restrict__ Wo16,
                         const float* __restrict__ bo,
                         float* __restrict__ y) {
  const int lane = threadIdx.x;
  const int tm = blockIdx.x, tc = blockIdx.y;
  const _Float16* arow = outs + (size_t)tm * 16 * DIMN;
  const _Float16* bcol = Wo16 + (size_t)tc * 16 * DIMN;
  v8f acc = {};
#pragma unroll
  for (int k0 = 0; k0 < DIMN; k0 += 32)
    acc = wmma16(load_a16(arow, DIMN, k0, lane), load_b16(bcol, DIMN, k0, lane), acc);

  const int half = lane >> 4;
  const int c = tc * 16 + (lane & 15);
  const float bias = bo[c];
#pragma unroll
  for (int r = 0; r < 8; ++r) {
    const int node = tm * 16 + r + half * 8;
    y[(size_t)node * DIMN + c] = acc[r] + bias;
  }
}

// ---------------------------------------------------------------------------
// 6) Edge gather: ea[h, b*E + j] = P[b,h,src,tgt].
// ---------------------------------------------------------------------------
__global__ void gla_gather(const float* __restrict__ P,
                           const int* __restrict__ ei,
                           float* __restrict__ ea) {
  const int idx = blockIdx.x * 256 + threadIdx.x;   // 0..131071 == b*E + j
  const int b = idx >> 11, j = idx & (EN - 1);
  const int s = ei[(size_t)b * 2 * EN + j];
  const int t = ei[(size_t)b * 2 * EN + EN + j];
  const float* pb = P + ((size_t)(b * HN) * NPGN + s) * NPGN + t;
#pragma unroll
  for (int h = 0; h < HN; ++h)
    ea[(size_t)h * (BG * EN) + idx] = pb[(size_t)h * NPGN * NPGN];
}

// ---------------------------------------------------------------------------
extern "C" void kernel_launch(void* const* d_in, const int* in_sizes, int n_in,
                              void* d_out, int out_size, void* d_ws, size_t ws_size,
                              hipStream_t stream) {
  (void)in_sizes; (void)n_in; (void)out_size; (void)ws_size;
  const float* x  = (const float*)d_in[0];
  const int*   ei = (const int*)d_in[1];
  const float* Wq = (const float*)d_in[2];
  const float* Wk = (const float*)d_in[3];
  const float* Wv = (const float*)d_in[4];
  const float* We = (const float*)d_in[5];
  const float* Ws = (const float*)d_in[6];
  const float* Wo = (const float*)d_in[7];
  const float* bo = (const float*)d_in[8];
  float* out = (float*)d_out;
  float* ea  = out + (size_t)NODES * DIMN;

  char* ws = (char*)d_ws;
  size_t off = 0;
  auto alloc = [&](size_t bytes) -> void* {
    void* p = ws + off;
    off += (bytes + 255) & ~(size_t)255;
    return p;
  };
  _Float16* xb   = (_Float16*)alloc((size_t)NODES * DIMN * 2);     // 16 MB
  _Float16* Wq16 = (_Float16*)alloc((size_t)DIMN * DIMN * 2);
  _Float16* Wk16 = (_Float16*)alloc((size_t)DIMN * DIMN * 2);
  _Float16* Wv16 = (_Float16*)alloc((size_t)DIMN * DIMN * 2);
  _Float16* Wo16 = (_Float16*)alloc((size_t)DIMN * DIMN * 2);
  _Float16* qb   = (_Float16*)alloc((size_t)NODES * DIMN * 2);     // [b,h,n,d]
  _Float16* kbuf = (_Float16*)alloc((size_t)NODES * DIMN * 2);     // [b,h,n,d] (scaled)
  _Float16* vT   = (_Float16*)alloc((size_t)NODES * DIMN * 2);     // [b,h,d,n]
  float*    vf   = (float*)   alloc((size_t)NODES * DIMN * 4);     // [b,h,n,d]
  float*    P    = (float*)   alloc((size_t)BG * HN * NPGN * NPGN * 4); // 128 MB
  _Float16* outs = (_Float16*)alloc((size_t)NODES * DIMN * 2);     // [node][c]

  // 0) conversions
  gla_cvt_f16<<<(NODES * DIMN + 255) / 256, 256, 0, stream>>>(x, xb, NODES * DIMN);
  gla_cvt_f16<<<(DIMN * DIMN + 255) / 256, 256, 0, stream>>>(Wq, Wq16, DIMN * DIMN);
  gla_cvt_f16<<<(DIMN * DIMN + 255) / 256, 256, 0, stream>>>(Wk, Wk16, DIMN * DIMN);
  gla_cvt_f16<<<(DIMN * DIMN + 255) / 256, 256, 0, stream>>>(Wv, Wv16, DIMN * DIMN);
  gla_cvt_f16<<<(DIMN * DIMN + 255) / 256, 256, 0, stream>>>(Wo, Wo16, DIMN * DIMN);

  // 1) q/k/v projections (WMMA)
  gla_qkv_gemm<<<dim3(NODES / 16, DIMN / 16, 3), 32, 0, stream>>>(
      xb, Wq16, Wk16, Wv16, qb, kbuf, vT, vf);

  // 2) scores + head-mix + GELU (WMMA + VALU)
  gla_scores<<<dim3(16, 16, BG), 32, 0, stream>>>(qb, kbuf, We, Ws, P);

  // 3) softmax rows
  gla_softmax<<<(BG * HN * NPGN) / 8, dim3(32, 8), 0, stream>>>(P);

  // 4) v - A v (WMMA)
  gla_av<<<dim3(DH / 16, NPGN / 16, BG * HN), 32, 0, stream>>>(P, vT, vf, outs);

  // 5) output projection + bias (WMMA)
  gla_proj<<<dim3(NODES / 16, DIMN / 16), 32, 0, stream>>>(outs, Wo16, bo, out);

  // 6) edge gather
  gla_gather<<<(BG * EN) / 256, 256, 0, stream>>>(P, ei, ea);
}
